// VAE_65369402245819
// MI455X (gfx1250) — compile-verified, matrix operand order checked by reading, not run on hardware
//
#include <hip/hip_runtime.h>
#include <hip/hip_bf16.h>
#include <math.h>

// ---------------------------------------------------------------------------
// Problem constants (match reference)
// ---------------------------------------------------------------------------
constexpr int kV  = 30000;
constexpr int kH1 = 1024;
constexpr int kH2 = 512;
constexpr int kL  = 200;
constexpr int kB  = 128;
constexpr int kS  = 50;
constexpr int kBS = kB * kS;           // 6400 rows for sentiment MLP

typedef __attribute__((ext_vector_type(16))) __bf16 v16bf;
typedef __attribute__((ext_vector_type(8)))  __bf16 v8bf;
typedef __attribute__((ext_vector_type(4)))  __bf16 v4bf;
typedef __attribute__((ext_vector_type(8)))  float  v8f;
typedef __attribute__((ext_vector_type(4)))  float  v4f;

// ---------------------------------------------------------------------------
// WMMA GEMM:  C[M,N] = act(A[M,K] @ W[K,N] + bias[N])
//   256 threads = 8 waves; block tile 64(M) x 64(N); K step 32.
//   Waves 4(M) x 2(N); each wave: 16x32 output, one A fragment reused across
//   two v_wmma_f32_16x16x32_bf16; fragments = 2x ds_load_b128 each.
//   Staging vectorized (float4 global loads, b64 LDS stores for A) when
//   K%4==0 && N%4==0; guarded scalar fallback otherwise (K=201 case).
//   act: 0 = none, 1 = relu, 2 = sigmoid
// ---------------------------------------------------------------------------
#define TM 64
#define TN 64
#define TK 32
#define LDK (TK + 8)   // 40 halves = 80 bytes row pitch (16B and 8B multiple)

__global__ __launch_bounds__(256)
void k_gemm(const float* __restrict__ A, const float* __restrict__ W,
            const float* __restrict__ bias, float* __restrict__ C,
            int M, int N, int K, int act)
{
    __shared__ __bf16 sA [TM][LDK];   // [m][k]
    __shared__ __bf16 sBt[TN][LDK];   // [n][k]  (transposed B tile)

    const int tid  = threadIdx.x;
    const int lane = tid & 31;
    const int wave = tid >> 5;          // 0..7
    const int wm   = wave >> 1;         // 0..3  (M sub-tile)
    const int wn   = wave & 1;          // 0..1  (N sub-tile pair)

    const int m0 = blockIdx.x * TM;
    const int n0 = blockIdx.y * TN;

    v8f acc0 = {};
    v8f acc1 = {};

    const int half = lane >> 4;         // K-half selector per CDNA5 layout
    const int l15  = lane & 15;
    const int rowA  = (wm << 4) + l15;
    const int rowB0 = (wn << 5) + l15;
    const int rowB1 = rowB0 + 16;
    const int kb    = half << 3;

    const bool vec_ok = ((K & 3) == 0) && ((N & 3) == 0);  // uniform

    // vectorized-staging coordinates
    const int arow = tid >> 3;          // 0..31 (A row, +32 second pass)
    const int ac4  = (tid & 7) << 2;    // A col group (0,4,..,28)
    const int bk   = tid >> 4;          // 0..15 (B k row, +16 second pass)
    const int bn4  = (tid & 15) << 2;   // B n group (0,4,..,60)

    for (int k0 = 0; k0 < K; k0 += TK) {
        // prefetch next K tile into GL2 (global_prefetch_b8)
        if (k0 + TK < K) {
            int pm = m0 + (tid & (TM - 1));
            if (pm < M)
                __builtin_prefetch(&A[(size_t)pm * K + (k0 + TK)], 0, 1);
            int pk = k0 + TK + (tid >> 3);
            if (pk < K)
                __builtin_prefetch(&W[(size_t)pk * N + n0], 0, 1);
        }

        if (vec_ok) {
            // ---- A tile: 2 x (float4 load -> v4bf LDS store) per thread ----
#pragma unroll
            for (int it = 0; it < 2; ++it) {
                int r  = arow + (it << 5);
                int gm = m0 + r, gk = k0 + ac4;
                v4bf o;
                if (gm < M && gk + 3 < K) {
                    v4f f = *(const v4f*)&A[(size_t)gm * K + gk];
                    o[0] = (__bf16)f[0]; o[1] = (__bf16)f[1];
                    o[2] = (__bf16)f[2]; o[3] = (__bf16)f[3];
                } else {
#pragma unroll
                    for (int j = 0; j < 4; ++j) {
                        float v = (gm < M && gk + j < K)
                                      ? A[(size_t)gm * K + gk + j] : 0.0f;
                        o[j] = (__bf16)v;
                    }
                }
                *(v4bf*)&sA[r][ac4] = o;   // 8B-aligned -> ds_store_b64
            }
            // ---- B tile: 2 x (float4 load along N -> 4 b16 transposed) ----
#pragma unroll
            for (int it = 0; it < 2; ++it) {
                int r  = bk + (it << 4);          // local k
                int gk = k0 + r, gn = n0 + bn4;
                __bf16 o0, o1, o2, o3;
                if (gk < K && gn + 3 < N) {
                    v4f f = *(const v4f*)&W[(size_t)gk * N + gn];
                    o0 = (__bf16)f[0]; o1 = (__bf16)f[1];
                    o2 = (__bf16)f[2]; o3 = (__bf16)f[3];
                } else {
                    float f0 = (gk < K && gn + 0 < N) ? W[(size_t)gk * N + gn + 0] : 0.0f;
                    float f1 = (gk < K && gn + 1 < N) ? W[(size_t)gk * N + gn + 1] : 0.0f;
                    float f2 = (gk < K && gn + 2 < N) ? W[(size_t)gk * N + gn + 2] : 0.0f;
                    float f3 = (gk < K && gn + 3 < N) ? W[(size_t)gk * N + gn + 3] : 0.0f;
                    o0 = (__bf16)f0; o1 = (__bf16)f1; o2 = (__bf16)f2; o3 = (__bf16)f3;
                }
                sBt[bn4 + 0][r] = o0;
                sBt[bn4 + 1][r] = o1;
                sBt[bn4 + 2][r] = o2;
                sBt[bn4 + 3][r] = o3;
            }
        } else {
            // ---- guarded scalar staging (odd-K case, e.g. K=201) ----
            for (int i = tid; i < TM * TK; i += 256) {
                int r = i >> 5, c = i & (TK - 1);
                int gm = m0 + r, gk = k0 + c;
                float v = (gm < M && gk < K) ? A[(size_t)gm * K + gk] : 0.0f;
                sA[r][c] = (__bf16)v;
            }
            for (int i = tid; i < TK * TN; i += 256) {
                int r = i >> 6, c = i & (TN - 1);
                int gk = k0 + r, gn = n0 + c;
                float v = (gk < K && gn < N) ? W[(size_t)gk * N + gn] : 0.0f;
                sBt[c][r] = (__bf16)v;
            }
        }
        __syncthreads();

        // Fragments: 2x ds_load_b128 each (pitch 80B, offsets 0/16/32/48)
        v16bf fa, fb0, fb1;
        ((v8bf*)&fa)[0]  = *(const v8bf*)&sA[rowA][kb];
        ((v8bf*)&fa)[1]  = *(const v8bf*)&sA[rowA][kb + 16];
        ((v8bf*)&fb0)[0] = *(const v8bf*)&sBt[rowB0][kb];
        ((v8bf*)&fb0)[1] = *(const v8bf*)&sBt[rowB0][kb + 16];
        ((v8bf*)&fb1)[0] = *(const v8bf*)&sBt[rowB1][kb];
        ((v8bf*)&fb1)[1] = *(const v8bf*)&sBt[rowB1][kb + 16];

        acc0 = __builtin_amdgcn_wmma_f32_16x16x32_bf16(
                   false, fa, false, fb0, (short)0, acc0, false, false);
        acc1 = __builtin_amdgcn_wmma_f32_16x16x32_bf16(
                   false, fa, false, fb1, (short)0, acc1, false, false);
        __syncthreads();
    }

    // Epilogue: C/D layout — VGPR r holds M = r + 8*(lane>=16), N = lane&15
#pragma unroll
    for (int sub = 0; sub < 2; ++sub) {
        const v8f accv = sub ? acc1 : acc0;
        const int n = n0 + (wn << 5) + (sub << 4) + l15;
        if (n < N) {
            float bv = bias ? bias[n] : 0.0f;
#pragma unroll
            for (int r = 0; r < 8; ++r) {
                int m = m0 + (wm << 4) + (half << 3) + r;
                if (m < M) {
                    float v = accv[r] + bv;
                    if (act == 1)      v = fmaxf(v, 0.0f);
                    else if (act == 2) v = 1.0f / (1.0f + __expf(-v));
                    C[(size_t)m * N + n] = v;
                }
            }
        }
    }
}

// ---------------------------------------------------------------------------
// Sentiment MLP layer 1 (K=1, rank-1 -> elementwise is optimal)
// ---------------------------------------------------------------------------
__global__ void k_sent_l1(const float* __restrict__ scores,
                          const float* __restrict__ w1,
                          const float* __restrict__ b1,
                          float* __restrict__ out, int rows, int H)
{
    size_t i = (size_t)blockIdx.x * blockDim.x + threadIdx.x;
    size_t total = (size_t)rows * H;
    if (i >= total) return;
    int r = (int)(i / H), j = (int)(i % H);
    float v = scores[r] * w1[j] + b1[j];
    out[i] = fmaxf(v, 0.0f);
}

// ---------------------------------------------------------------------------
// Row-weight reduction + cz update. One block per batch row b.
// ---------------------------------------------------------------------------
__global__ void k_roww(const float* __restrict__ swv,     // [B,S,L]
                       const float* __restrict__ scores,  // [B,S]
                       const int*   __restrict__ lengths, // [B]
                       float* __restrict__ cz, int lvl)   // [B,L]
{
    __shared__ float a[kS];
    const int b   = blockIdx.x;
    const int len = lengths[b];
    const float sF = 1.0f + 0.5f * (float)lvl;

    for (int s = threadIdx.x; s < kS; s += blockDim.x) {
        float sc = scores[b * kS + s];
        float v  = 1.0f;
        if (lvl == 1)      v = fabsf(sc);
        else if (lvl == 2) v = fabsf(sc - 0.5f) * 2.0f;
        a[s] = v;
    }
    __syncthreads();
    if (threadIdx.x == 0) {
        if (lvl == 0) {
            float inv = 1.0f / fmaxf((float)len, 1.0f);
            for (int s = 0; s < kS; ++s) a[s] = (s < len) ? inv : 0.0f;
        } else {
            float mx = -1e30f;
            for (int s = 0; s < len; ++s) mx = fmaxf(mx, a[s]);
            float sum = 0.0f;
            for (int s = 0; s < len; ++s) { a[s] = __expf(a[s] - mx); sum += a[s]; }
            float inv = (sum > 0.0f) ? 1.0f / sum : 0.0f;
            for (int s = 0; s < kS; ++s) a[s] = (s < len) ? a[s] * inv : 0.0f;
        }
    }
    __syncthreads();
    for (int l = threadIdx.x; l < kL; l += blockDim.x) {
        float rw = 0.0f;
#pragma unroll 5
        for (int s = 0; s < kS; ++s)
            rw += swv[((size_t)b * kS + s) * kL + l] * a[s];
        if (len == 0) rw = 1.0f;
        cz[b * kL + l] *= (1.0f + (rw - 0.5f) * sF);
    }
}

// ---------------------------------------------------------------------------
// Reparameterization: z = mu + exp(0.5*lv) * N(0,1) via hash Box-Muller
// ---------------------------------------------------------------------------
__device__ __forceinline__ unsigned hashu(unsigned x)
{
    x ^= x >> 16; x *= 0x7feb352du;
    x ^= x >> 15; x *= 0x846ca68bu;
    x ^= x >> 16; return x;
}

__global__ void k_reparam(const float* __restrict__ mu,
                          const float* __restrict__ lv,
                          float* __restrict__ z, int n, unsigned seed)
{
    int i = blockIdx.x * blockDim.x + threadIdx.x;
    if (i >= n) return;
    unsigned h1 = hashu((unsigned)i * 2654435761u + seed);
    unsigned h2 = hashu(h1 ^ 0x9e3779b9u);
    float u1 = ((float)h1 + 0.5f) * (1.0f / 4294967296.0f);
    float u2 = ((float)h2 + 0.5f) * (1.0f / 4294967296.0f);
    float g  = sqrtf(-2.0f * __logf(u1)) * __cosf(6.28318530717958647f * u2);
    z[i] = mu[i] + __expf(0.5f * lv[i]) * g;
}

// ---------------------------------------------------------------------------
// Build final-encoder input: fin[b,0:200] = cz[b,:], fin[b,200] = mean sentiment
// ---------------------------------------------------------------------------
__global__ void k_make_fin(const float* __restrict__ cz,
                           const float* __restrict__ scores,
                           const int*   __restrict__ lengths,
                           float* __restrict__ fin)
{
    const int b = blockIdx.x;
    __shared__ float sent;
    if (threadIdx.x == 0) {
        int len = lengths[b];
        float sum = 0.0f;
        for (int s = 0; s < len; ++s) sum += scores[b * kS + s];
        sent = (len == 0) ? 0.5f : sum / (float)len;
    }
    __syncthreads();
    for (int j = threadIdx.x; j < kL; j += blockDim.x)
        fin[b * (kL + 1) + j] = cz[b * kL + j];
    if (threadIdx.x == 0) fin[b * (kL + 1) + kL] = sent;
}

// ---------------------------------------------------------------------------
// Host-side orchestration
// ---------------------------------------------------------------------------
extern "C" void kernel_launch(void* const* d_in, const int* in_sizes, int n_in,
                              void* d_out, int out_size, void* d_ws, size_t ws_size,
                              hipStream_t stream)
{
    (void)in_sizes; (void)n_in; (void)out_size; (void)ws_size;

    const float* tfIdf  = (const float*)d_in[0];
    const float* scores = (const float*)d_in[1];
    const int*   lens   = (const int*)  d_in[2];
    const float* sw1 = (const float*)d_in[3];   const float* sb1 = (const float*)d_in[4];
    const float* sw2 = (const float*)d_in[5];   const float* sb2 = (const float*)d_in[6];
    const float* sw3 = (const float*)d_in[7];   const float* sb3 = (const float*)d_in[8];
    const float* bw1 = (const float*)d_in[9];   const float* bb1 = (const float*)d_in[10];
    const float* bw2 = (const float*)d_in[11];  const float* bb2 = (const float*)d_in[12];
    const float* bmw = (const float*)d_in[13];  const float* bmb = (const float*)d_in[14];
    const float* blw = (const float*)d_in[15];  const float* blb = (const float*)d_in[16];
    const float* rw1 = (const float*)d_in[17];  const float* rb1 = (const float*)d_in[18];
    const float* rw2 = (const float*)d_in[19];  const float* rb2 = (const float*)d_in[20];
    const float* rmw = (const float*)d_in[21];  const float* rmb = (const float*)d_in[22];
    const float* rlw = (const float*)d_in[23];  const float* rlb = (const float*)d_in[24];
    const float* fw1 = (const float*)d_in[25];  const float* fb1 = (const float*)d_in[26];
    const float* fw2 = (const float*)d_in[27];  const float* fb2 = (const float*)d_in[28];
    const float* fmw = (const float*)d_in[29];  const float* fmb = (const float*)d_in[30];
    const float* flw = (const float*)d_in[31];  const float* flb = (const float*)d_in[32];
    const float* dw1 = (const float*)d_in[33];  const float* db1 = (const float*)d_in[34];
    const float* dw2 = (const float*)d_in[35];  const float* db2 = (const float*)d_in[36];
    const float* dw3 = (const float*)d_in[37];  const float* db3 = (const float*)d_in[38];

    float* out = (float*)d_out;
    const size_t reconSz = (size_t)kB * kV;
    auto muP = [&](int i) { return out + reconSz + (size_t)i * kB * kL; };
    auto lvP = [&](int i) { return out + reconSz + (size_t)5 * kB * kL + (size_t)i * kB * kL; };

    float* wsf   = (float*)d_ws;
    float* s1out = wsf;                         // [6400,1024]
    float* s2out = s1out + (size_t)kBS * kH1;   // [6400, 512]
    float* swb   = s2out + (size_t)kBS * kH2;   // [6400, 200]
    float* hA    = swb   + (size_t)kBS * kL;    // [128,1024]
    float* hB    = hA    + (size_t)kB  * kH1;   // [128, 512]
    float* cz    = hB    + (size_t)kB  * kH2;   // [128, 200]
    float* fin   = cz    + (size_t)kB  * kL;    // [128, 201]
    float* fz    = fin   + (size_t)kB  * (kL+1);// [128, 200]

    auto gemm = [&](const float* A, const float* W, const float* bias, float* C,
                    int M, int N, int K, int act) {
        dim3 grid((M + TM - 1) / TM, (N + TN - 1) / TN);
        k_gemm<<<grid, 256, 0, stream>>>(A, W, bias, C, M, N, K, act);
    };
    auto reparam = [&](const float* mu, const float* lv, float* z, unsigned seed) {
        int n = kB * kL;
        k_reparam<<<(n + 255) / 256, 256, 0, stream>>>(mu, lv, z, n, seed);
    };

    // ---- base encoder ----
    gemm(tfIdf, bw1, bb1, hA, kB, kH1, kV, 1);          // dominant GEMM #1
    gemm(hA, bw2, bb2, hB, kB, kH2, kH1, 1);
    gemm(hB, bmw, bmb, muP(0), kB, kL, kH2, 0);
    gemm(hB, blw, blb, lvP(0), kB, kL, kH2, 0);
    reparam(muP(0), lvP(0), cz, 0x1234u);

    // ---- three sentiment-weighted recursive levels ----
    for (int lvl = 0; lvl < 3; ++lvl) {
        size_t t1 = (size_t)kBS * kH1;
        k_sent_l1<<<(unsigned)((t1 + 255) / 256), 256, 0, stream>>>(
            scores, sw1 + (size_t)lvl * kH1, sb1 + (size_t)lvl * kH1, s1out, kBS, kH1);
        gemm(s1out, sw2 + (size_t)lvl * kH1 * kH2, sb2 + (size_t)lvl * kH2,
             s2out, kBS, kH2, kH1, 1);                  // bulk sentiment GEMM
        gemm(s2out, sw3 + (size_t)lvl * kH2 * kL, sb3 + (size_t)lvl * kL,
             swb, kBS, kL, kH2, 2);                     // sigmoid head
        k_roww<<<kB, 256, 0, stream>>>(swb, scores, lens, cz, lvl);
        gemm(cz, rw1, rb1, hA, kB, kH1, kL, 1);
        gemm(hA, rw2, rb2, hB, kB, kH2, kH1, 1);
        gemm(hB, rmw, rmb, muP(1 + lvl), kB, kL, kH2, 0);
        gemm(hB, rlw, rlb, lvP(1 + lvl), kB, kL, kH2, 0);
        reparam(muP(1 + lvl), lvP(1 + lvl), cz, 0x5678u + (unsigned)lvl);
    }

    // ---- final encoder (latent + mean sentiment) ----
    k_make_fin<<<kB, 256, 0, stream>>>(cz, scores, lens, fin);
    gemm(fin, fw1, fb1, hA, kB, kH1, kL + 1, 1);        // K=201 -> scalar staging
    gemm(hA, fw2, fb2, hB, kB, kH2, kH1, 1);
    gemm(hB, fmw, fmb, muP(4), kB, kL, kH2, 0);
    gemm(hB, flw, flb, lvP(4), kB, kL, kH2, 0);
    reparam(muP(4), lvP(4), fz, 0x9abcu);

    // ---- decoder ----
    gemm(fz, dw1, db1, hB, kB, kH2, kL, 1);
    gemm(hB, dw2, db2, hA, kB, kH1, kH2, 1);
    gemm(hA, dw3, db3, out, kB, kV, kH1, 2);            // dominant GEMM #2 (sigmoid)
}
